// Encoder_1297080123914
// MI455X (gfx1250) — compile-verified, matrix operand order checked by reading, not run on hardware
//
#include <hip/hip_runtime.h>
#include <hip/hip_bf16.h>

// ---------------------------------------------------------------------------
// CDNA5 (gfx1250) transformer encoder forward.
// f16 data path (converted once), v_wmma_f32_16x16x32_f16 matmuls,
// async global->LDS tile copies (GLOBAL_LOAD_ASYNC_TO_LDS_B128 / ASYNCcnt).
// ---------------------------------------------------------------------------

typedef __attribute__((ext_vector_type(16))) _Float16 v16h;
typedef __attribute__((ext_vector_type(8)))  float    v8f;
typedef int v4i_vec __attribute__((vector_size(16)));   // int __vector(4)

constexpr int E_   = 512;
constexpr int H_   = 8;
constexpr int D_   = 64;
constexpr int FF_  = 2048;
constexpr int L_   = 2;
constexpr int B_   = 4;
constexpr int S_   = 2048;
constexpr int ROWS = B_ * S_;          // 8192 token rows

#if defined(__has_builtin)
#if __has_builtin(__builtin_amdgcn_global_load_async_to_lds_b128)
#define HAVE_ASYNC_LDS 1
#endif
#endif
#ifndef HAVE_ASYNC_LDS
#define HAVE_ASYNC_LDS 0
#endif

__device__ __forceinline__ uint32_t pack2h(float a, float b) {
  union { _Float16 h[2]; uint32_t u; } t;
  t.h[0] = (_Float16)a; t.h[1] = (_Float16)b;
  return t.u;
}

// 16-byte global -> LDS copy. Async (ASYNCcnt) when available.
__device__ __forceinline__ void cp16B_g2l(const _Float16* g, _Float16* l) {
#if HAVE_ASYNC_LDS
  __builtin_amdgcn_global_load_async_to_lds_b128(
      (__attribute__((address_space(1))) v4i_vec*)(_Float16*)g,
      (__attribute__((address_space(3))) v4i_vec*)l, 0, 0);
#else
  *(uint4*)l = *(const uint4*)g;
#endif
}

__device__ __forceinline__ void async_fence() {
#if HAVE_ASYNC_LDS
  asm volatile("s_wait_asynccnt 0" ::: "memory");
#endif
}

// Load one 16x32 f16 WMMA operand fragment from LDS (A/B layout per ISA):
//   lane L: row = L & 15 ; k-base = (L>=16) ? 8 : 0
//   VGPR v (0..3): k = kbase + 2v ; VGPR v (4..7): k = kbase + 16 + 2(v-4)
__device__ __forceinline__ v16h load_frag_k32(const _Float16* base, int rowStride, int lane) {
  int r  = lane & 15;
  int kh = (lane >> 4) << 3;
  const _Float16* p = base + r * rowStride + kh;
  union { v16h v; uint32_t u[8]; } f;
  const uint32_t* p0 = (const uint32_t*)p;
  const uint32_t* p1 = (const uint32_t*)(p + 16);
#pragma unroll
  for (int i = 0; i < 4; ++i) { f.u[i] = p0[i]; f.u[4 + i] = p1[i]; }
  return f.v;
}

__device__ __forceinline__ v8f wmma_f16(v16h a, v16h b, v8f c) {
  return __builtin_amdgcn_wmma_f32_16x16x32_f16(false, a, false, b, (short)0, c, false, false);
}

// ---------------------------------------------------------------------------
// Weight convert + transpose: W[K][N] f32 -> Wt[N][K] f16 (one-time, tiny)
// ---------------------------------------------------------------------------
__global__ __launch_bounds__(256) void wconv_kernel(
    const float* __restrict__ W, _Float16* __restrict__ Wt, int K, int N) {
  int idx = blockIdx.x * 256 + threadIdx.x;   // over N*K
  int n = idx / K;
  int k = idx - n * K;
  Wt[idx] = (_Float16)W[(size_t)k * N + n];
}

// ---------------------------------------------------------------------------
// Embedding + positional add: writes f32 h and f16 h16
// ---------------------------------------------------------------------------
__global__ __launch_bounds__(128) void embed_kernel(
    const int* __restrict__ x, const float* __restrict__ emb,
    const float* __restrict__ pos, float* __restrict__ h,
    _Float16* __restrict__ h16) {
  int row = blockIdx.x;
  int s   = row & (S_ - 1);
  int tok = x[row] + 1;
  int c   = threadIdx.x << 2;
  float4 e4 = *(const float4*)(emb + (size_t)tok * E_ + c);
  float4 p4 = *(const float4*)(pos + (size_t)s   * E_ + c);
  float4 o; o.x = e4.x + p4.x; o.y = e4.y + p4.y; o.z = e4.z + p4.z; o.w = e4.w + p4.w;
  *(float4*)(h + (size_t)row * E_ + c) = o;
  uint2 hp; hp.x = pack2h(o.x, o.y); hp.y = pack2h(o.z, o.w);
  *(uint2*)(h16 + (size_t)row * E_ + c) = hp;
}

// ---------------------------------------------------------------------------
// WMMA GEMM, f16 in: C[M,N] = act(A[M,K] @ Wt[N,K]^T + bias[N]), M = 8192.
// Block tile 64x256, 8 waves (2x4), wave tile 32x64, K-step 32, dbl buffer.
// Output: f32 (Cf) and/or f16 (Ch).
// ---------------------------------------------------------------------------
__global__ __launch_bounds__(256) void gemm_f16_kernel(
    const _Float16* __restrict__ A, const _Float16* __restrict__ Wt,
    const float* __restrict__ bias, float* __restrict__ Cf,
    _Float16* __restrict__ Ch, int N, int Kdim, int relu) {
  __shared__ _Float16 Atile[2][64][32];        //  8 KB
  __shared__ _Float16 Btile[2][256][32];       // 32 KB  ([n][k] fragment order)

  const int tid  = threadIdx.x;
  const int lane = tid & 31;
  const int wave = tid >> 5;
  const int wm   = (wave >> 2) << 5;           // 0 / 32
  const int wn   = (wave & 3) << 6;            // 0 / 64 / 128 / 192
  const int m0   = blockIdx.x * 64;
  const int n0   = blockIdx.y * 256;

  v8f acc[2][4] = {};
  const int nk = Kdim >> 5;

  auto load_tiles = [&](int kt, int buf) {
    const int k0 = kt << 5;
    {  // A: 64x32 f16 = 256 x 16B, 1 per thread
      int r = tid >> 2, seg = (tid & 3) << 3;
      cp16B_g2l(A + (size_t)(m0 + r) * Kdim + k0 + seg, &Atile[buf][r][seg]);
    }
#pragma unroll
    for (int t = 0; t < 4; ++t) {  // B: 256x32 f16 = 1024 x 16B, 4 per thread
      int idx = t * 256 + tid;
      int r = idx >> 2, seg = (idx & 3) << 3;
      cp16B_g2l(Wt + (size_t)(n0 + r) * Kdim + k0 + seg, &Btile[buf][r][seg]);
    }
  };

  load_tiles(0, 0);
  async_fence();
  __syncthreads();

  for (int kt = 0; kt < nk; ++kt) {
    const int cur = kt & 1;
    if (kt + 1 < nk) load_tiles(kt + 1, cur ^ 1);

    v16h af[2], bf[4];
    af[0] = load_frag_k32(&Atile[cur][wm + 0][0], 32, lane);
    af[1] = load_frag_k32(&Atile[cur][wm + 16][0], 32, lane);
#pragma unroll
    for (int j = 0; j < 4; ++j)
      bf[j] = load_frag_k32(&Btile[cur][wn + j * 16][0], 32, lane);
#pragma unroll
    for (int i = 0; i < 2; ++i)
#pragma unroll
      for (int j = 0; j < 4; ++j)
        acc[i][j] = wmma_f16(af[i], bf[j], acc[i][j]);

    async_fence();
    __syncthreads();
  }

  const int lrow = (lane >> 4) << 3;
  const int lcol = lane & 15;
#pragma unroll
  for (int i = 0; i < 2; ++i)
#pragma unroll
    for (int j = 0; j < 4; ++j) {
      int col = n0 + wn + j * 16 + lcol;
      float bv = bias[col];
#pragma unroll
      for (int e = 0; e < 8; ++e) {
        int row = m0 + wm + i * 16 + e + lrow;
        float v = acc[i][j][e] + bv;
        if (relu) v = fmaxf(v, 0.0f);
        if (Cf) Cf[(size_t)row * N + col] = v;
        if (Ch) Ch[(size_t)row * N + col] = (_Float16)v;
      }
    }
}

// ---------------------------------------------------------------------------
// Flash attention (f16 Q/K/V in, f16 out). One block = (b,h) x 256 q-rows.
// ---------------------------------------------------------------------------
__global__ __launch_bounds__(256) void attention_kernel(
    const _Float16* __restrict__ Q, const _Float16* __restrict__ K,
    const _Float16* __restrict__ V, const int* __restrict__ mask,
    _Float16* __restrict__ O) {
  __shared__ _Float16 Kt[64][64];         //  8 KB : [key][d]  (B-frag for QK^T)
  __shared__ _Float16 Vt[64][64];         //  8 KB : [d][key]  (B-frag for P*V)
  __shared__ _Float16 Pbuf[8][32][64];    // 32 KB : per-wave P tile
  __shared__ int      smask[64];

  const int tid  = threadIdx.x;
  const int lane = tid & 31;
  const int wave = tid >> 5;
  const int bh   = blockIdx.x;
  const int b    = bh >> 3;
  const int hh   = bh & 7;
  const int q0   = blockIdx.y * 256;
  const size_t base = ((size_t)b * S_) * E_ + (size_t)hh * D_;

  // Q fragments straight from global f16 (A layout)
  v16h qf[2][2];
#pragma unroll
  for (int i = 0; i < 2; ++i)
#pragma unroll
    for (int s = 0; s < 2; ++s) {
      int q = q0 + wave * 32 + i * 16 + (lane & 15);
      const _Float16* rp = Q + base + (size_t)q * E_ + s * 32 + ((lane >> 4) << 3);
      union { v16h v; uint32_t u[8]; } f;
#pragma unroll
      for (int vv = 0; vv < 8; ++vv) {
        int d = ((vv >> 2) << 4) + ((vv & 3) << 1);
        f.u[vv] = *(const uint32_t*)(rp + d);
      }
      qf[i][s] = f.v;
    }

  v8f o[2][4] = {};
  float mst[2][8], lst[2][8];
#pragma unroll
  for (int i = 0; i < 2; ++i)
#pragma unroll
    for (int e = 0; e < 8; ++e) { mst[i][e] = -1e30f; lst[i][e] = 0.0f; }

  const float scale = 0.125f;
  const int lrow = (lane >> 4) << 3;
  const int lcol = lane & 15;

  for (int kb = 0; kb < S_; kb += 64) {
    __syncthreads();
    // K block: [key][d] f16 rows -> direct (async) copy; V -> transposed stores
#pragma unroll
    for (int t = 0; t < 2; ++t) {
      int idx = t * 256 + tid;
      int r = idx >> 3, c0 = (idx & 7) << 3;
      cp16B_g2l(K + base + (size_t)(kb + r) * E_ + c0, &Kt[r][c0]);
      uint4 vv = *(const uint4*)(V + base + (size_t)(kb + r) * E_ + c0);
      const _Float16* vh = (const _Float16*)&vv;
#pragma unroll
      for (int j = 0; j < 8; ++j) Vt[c0 + j][r] = vh[j];
    }
    if (tid < 64) smask[tid] = mask[(size_t)b * S_ + kb + tid];
    async_fence();
    __syncthreads();

    // scores S = Q K^T
    v16h kf[4][2];
#pragma unroll
    for (int j = 0; j < 4; ++j)
#pragma unroll
      for (int s = 0; s < 2; ++s)
        kf[j][s] = load_frag_k32(&Kt[j * 16][s * 32], 64, lane);

    v8f sa[2][4];
#pragma unroll
    for (int i = 0; i < 2; ++i)
#pragma unroll
      for (int j = 0; j < 4; ++j) {
        v8f z = {};
        z = wmma_f16(qf[i][0], kf[j][0], z);
        sa[i][j] = wmma_f16(qf[i][1], kf[j][1], z);
      }

    int mj[4];
#pragma unroll
    for (int j = 0; j < 4; ++j) mj[j] = smask[j * 16 + lcol];
#pragma unroll
    for (int i = 0; i < 2; ++i)
#pragma unroll
      for (int j = 0; j < 4; ++j)
#pragma unroll
        for (int e = 0; e < 8; ++e) {
          float v = sa[i][j][e] * scale;
          sa[i][j][e] = mj[j] ? v : -1e20f;
        }

    // online softmax per m-tile
#pragma unroll
    for (int i = 0; i < 2; ++i) {
      float nm[8];
#pragma unroll
      for (int e = 0; e < 8; ++e) {
        float v = sa[i][0][e];
        v = fmaxf(v, sa[i][1][e]); v = fmaxf(v, sa[i][2][e]); v = fmaxf(v, sa[i][3][e]);
        nm[e] = v;
      }
#pragma unroll
      for (int off = 8; off >= 1; off >>= 1)
#pragma unroll
        for (int e = 0; e < 8; ++e)
          nm[e] = fmaxf(nm[e], __shfl_xor(nm[e], off, 16));

      float fs[8];
#pragma unroll
      for (int e = 0; e < 8; ++e) {
        float mn = fmaxf(mst[i][e], nm[e]);
        fs[e] = __expf(mst[i][e] - mn);
        mst[i][e] = mn;
      }
#pragma unroll
      for (int j = 0; j < 4; ++j)
#pragma unroll
        for (int e = 0; e < 8; ++e)
          o[i][j][e] *= fs[e];

      float rs[8] = {0, 0, 0, 0, 0, 0, 0, 0};
#pragma unroll
      for (int j = 0; j < 4; ++j)
#pragma unroll
        for (int e = 0; e < 8; ++e) {
          float p = __expf(sa[i][j][e] - mst[i][e]);
          sa[i][j][e] = p;
          rs[e] += p;
        }
#pragma unroll
      for (int off = 8; off >= 1; off >>= 1)
#pragma unroll
        for (int e = 0; e < 8; ++e)
          rs[e] += __shfl_xor(rs[e], off, 16);
#pragma unroll
      for (int e = 0; e < 8; ++e)
        lst[i][e] = lst[i][e] * fs[e] + rs[e];

#pragma unroll
      for (int j = 0; j < 4; ++j)
#pragma unroll
        for (int e = 0; e < 8; ++e)
          Pbuf[wave][i * 16 + e + lrow][j * 16 + lcol] = (_Float16)sa[i][j][e];
    }

    asm volatile("s_wait_dscnt 0" ::: "memory");  // wave-local P round-trip

    v16h vf[4][2];
#pragma unroll
    for (int j = 0; j < 4; ++j)
#pragma unroll
      for (int s = 0; s < 2; ++s)
        vf[j][s] = load_frag_k32(&Vt[j * 16][s * 32], 64, lane);

#pragma unroll
    for (int i = 0; i < 2; ++i) {
      v16h pf0 = load_frag_k32(&Pbuf[wave][i * 16][0],  64, lane);
      v16h pf1 = load_frag_k32(&Pbuf[wave][i * 16][32], 64, lane);
#pragma unroll
      for (int j = 0; j < 4; ++j) {
        o[i][j] = wmma_f16(pf0, vf[j][0], o[i][j]);
        o[i][j] = wmma_f16(pf1, vf[j][1], o[i][j]);
      }
    }
  }

#pragma unroll
  for (int i = 0; i < 2; ++i)
#pragma unroll
    for (int j = 0; j < 4; ++j)
#pragma unroll
      for (int e = 0; e < 8; ++e) {
        int qrow = q0 + wave * 32 + i * 16 + e + lrow;
        int dcol = j * 16 + lcol;
        O[base + (size_t)qrow * E_ + dcol] = (_Float16)(o[i][j][e] / lst[i][e]);
      }
}

// ---------------------------------------------------------------------------
// Fused residual + LayerNorm: y = LN(x + r)*g + b ; writes f32 and f16 copies
// ---------------------------------------------------------------------------
__global__ __launch_bounds__(128) void add_ln_kernel(
    const float* __restrict__ X, const float* __restrict__ R,
    const float* __restrict__ g, const float* __restrict__ bb,
    float* __restrict__ Y, _Float16* __restrict__ Y16) {
  __shared__ float ssum[4], ssq[4];
  const int row  = blockIdx.x;
  const int tid  = threadIdx.x;
  const int lane = tid & 31;
  const int wave = tid >> 5;
  const int c0   = tid << 2;

  float4 x4 = *(const float4*)(X + (size_t)row * E_ + c0);
  float4 r4 = *(const float4*)(R + (size_t)row * E_ + c0);
  float v[4] = {x4.x + r4.x, x4.y + r4.y, x4.z + r4.z, x4.w + r4.w};
  float s = v[0] + v[1] + v[2] + v[3];
  float q = v[0]*v[0] + v[1]*v[1] + v[2]*v[2] + v[3]*v[3];
#pragma unroll
  for (int off = 16; off >= 1; off >>= 1) {
    s += __shfl_xor(s, off, 32);
    q += __shfl_xor(q, off, 32);
  }
  if (lane == 0) { ssum[wave] = s; ssq[wave] = q; }
  __syncthreads();
  float S1 = ssum[0] + ssum[1] + ssum[2] + ssum[3];
  float S2 = ssq[0]  + ssq[1]  + ssq[2]  + ssq[3];
  float mu   = S1 * (1.0f / E_);
  float var  = S2 * (1.0f / E_) - mu * mu;
  float rinv = rsqrtf(var + 1e-6f);
  float4 o;
  float* op = &o.x;
#pragma unroll
  for (int c = 0; c < 4; ++c) {
    int col = c0 + c;
    op[c] = (v[c] - mu) * rinv * g[col] + bb[col];
  }
  *(float4*)(Y + (size_t)row * E_ + c0) = o;
  uint2 hp; hp.x = pack2h(o.x, o.y); hp.y = pack2h(o.z, o.w);
  *(uint2*)(Y16 + (size_t)row * E_ + c0) = hp;
}

// ---------------------------------------------------------------------------
// Host orchestration
// ---------------------------------------------------------------------------
extern "C" void kernel_launch(void* const* d_in, const int* in_sizes, int n_in,
                              void* d_out, int out_size, void* d_ws, size_t ws_size,
                              hipStream_t stream) {
  const int*   x    = (const int*)d_in[0];
  const int*   mask = (const int*)d_in[1];
  const float* emb  = (const float*)d_in[2];
  const float* pos  = (const float*)d_in[3];
  const float* Wq   = (const float*)d_in[4];
  const float* bq   = (const float*)d_in[5];
  const float* Wk   = (const float*)d_in[6];
  const float* bk   = (const float*)d_in[7];
  const float* Wv   = (const float*)d_in[8];
  const float* bv   = (const float*)d_in[9];
  const float* Wo   = (const float*)d_in[10];
  const float* bo   = (const float*)d_in[11];
  const float* g1   = (const float*)d_in[12];
  const float* be1  = (const float*)d_in[13];
  const float* g2   = (const float*)d_in[14];
  const float* be2  = (const float*)d_in[15];
  const float* W1   = (const float*)d_in[16];
  const float* bf1  = (const float*)d_in[17];
  const float* W2   = (const float*)d_in[18];
  const float* bf2  = (const float*)d_in[19];

  float* h = (float*)d_out;                               // [B,S,E] f32, persistent

  const size_t ACT = (size_t)ROWS * E_;
  char* p = (char*)d_ws;
  auto alloc_h = [&](size_t n) { _Float16* r = (_Float16*)p; p += n * sizeof(_Float16); return r; };
  _Float16* h16   = alloc_h(ACT);
  _Float16* q16   = alloc_h(ACT);
  _Float16* k16   = alloc_h(ACT);
  _Float16* v16   = alloc_h(ACT);
  _Float16* att16 = alloc_h(ACT);
  _Float16* big16 = alloc_h((size_t)ROWS * FF_);
  float* f32tmp = (float*)p; p += ACT * sizeof(float);
  _Float16* wtq[L_]; _Float16* wtk[L_]; _Float16* wtv[L_]; _Float16* wto[L_];
  _Float16* wt1[L_]; _Float16* wt2[L_];
  for (int l = 0; l < L_; ++l) {
    wtq[l] = alloc_h((size_t)E_ * E_);
    wtk[l] = alloc_h((size_t)E_ * E_);
    wtv[l] = alloc_h((size_t)E_ * E_);
    wto[l] = alloc_h((size_t)E_ * E_);
    wt1[l] = alloc_h((size_t)E_ * FF_);
    wt2[l] = alloc_h((size_t)FF_ * E_);
  }

  // One-time weight convert+transpose (W[K][N] f32 -> Wt[N][K] f16)
  const int gEE = E_ * E_ / 256, gEF = E_ * FF_ / 256;
  for (int l = 0; l < L_; ++l) {
    wconv_kernel<<<gEE, 256, 0, stream>>>(Wq + (size_t)l * E_ * E_, wtq[l], E_, E_);
    wconv_kernel<<<gEE, 256, 0, stream>>>(Wk + (size_t)l * E_ * E_, wtk[l], E_, E_);
    wconv_kernel<<<gEE, 256, 0, stream>>>(Wv + (size_t)l * E_ * E_, wtv[l], E_, E_);
    wconv_kernel<<<gEE, 256, 0, stream>>>(Wo + (size_t)l * E_ * E_, wto[l], E_, E_);
    wconv_kernel<<<gEF, 256, 0, stream>>>(W1 + (size_t)l * E_ * FF_, wt1[l], E_, FF_);
    wconv_kernel<<<gEF, 256, 0, stream>>>(W2 + (size_t)l * FF_ * E_, wt2[l], FF_, E_);
  }

  embed_kernel<<<ROWS, 128, 0, stream>>>(x, emb, pos, h, h16);

  const dim3 gE(ROWS / 64, E_ / 256);                     // N = 512
  const dim3 gF(ROWS / 64, FF_ / 256);                    // N = 2048
  const dim3 gA(B_ * H_, S_ / 256);

  for (int l = 0; l < L_; ++l) {
    const float* bql = bq  + (size_t)l * E_;
    const float* bkl = bk  + (size_t)l * E_;
    const float* bvl = bv  + (size_t)l * E_;
    const float* bol = bo  + (size_t)l * E_;
    const float* b1l = bf1 + (size_t)l * FF_;
    const float* b2l = bf2 + (size_t)l * E_;
    const float* g1l = g1  + (size_t)l * E_;
    const float* e1l = be1 + (size_t)l * E_;
    const float* g2l = g2  + (size_t)l * E_;
    const float* e2l = be2 + (size_t)l * E_;

    gemm_f16_kernel<<<gE, 256, 0, stream>>>(h16, wtq[l], bql, nullptr, q16, E_, E_, 0);
    gemm_f16_kernel<<<gE, 256, 0, stream>>>(h16, wtk[l], bkl, nullptr, k16, E_, E_, 0);
    gemm_f16_kernel<<<gE, 256, 0, stream>>>(h16, wtv[l], bvl, nullptr, v16, E_, E_, 0);

    attention_kernel<<<gA, 256, 0, stream>>>(q16, k16, v16, mask, att16);

    gemm_f16_kernel<<<gE, 256, 0, stream>>>(att16, wto[l], bol, f32tmp, nullptr, E_, E_, 0);
    add_ln_kernel<<<ROWS, 128, 0, stream>>>(f32tmp, h, g1l, e1l, h, h16);

    gemm_f16_kernel<<<gF, 256, 0, stream>>>(h16, wt1[l], b1l, nullptr, big16, FF_, E_, 1);
    gemm_f16_kernel<<<gE, 256, 0, stream>>>(big16, wt2[l], b2l, f32tmp, nullptr, E_, FF_, 0);
    add_ln_kernel<<<ROWS, 128, 0, stream>>>(f32tmp, h, g2l, e2l, h, h16);
  }
}
